// NeuralModel_79370995630372
// MI455X (gfx1250) — matrix operand order repaired and checked by reading, not compiled, not run on hardware
//
#include <hip/hip_runtime.h>

typedef __attribute__((ext_vector_type(16))) _Float16 v16h;
typedef __attribute__((ext_vector_type(8)))  float    v8f;

union AV { v16h h; uint32_t u[8]; };

__device__ __forceinline__ float tanh_fast(float x) {
#if __has_builtin(__builtin_amdgcn_tanhf)
    return __builtin_amdgcn_tanhf(x);          // v_tanh_f32 (CDNA5 TRANS op)
#elif __has_builtin(__builtin_amdgcn_tanh_f32)
    return __builtin_amdgcn_tanh_f32(x);
#else
    float e = __expf(2.0f * x);                // v_exp_f32; saturates correctly at +-inf
    return 1.0f - 2.0f / (e + 1.0f);
#endif
}

__device__ __forceinline__ uint32_t pack2(float a, float b) {
    union { uint32_t u; _Float16 h[2]; } r;
    r.h[0] = (_Float16)a; r.h[1] = (_Float16)b;
    return r.u;
}

__device__ __forceinline__ uint32_t permx16_u(uint32_t v, int lane) {
    return (uint32_t)__builtin_amdgcn_ds_bpermute(((lane ^ 16) << 2), (int)v);
}
__device__ __forceinline__ float permx16_f(float v, int lane) {
    return __int_as_float(__builtin_amdgcn_ds_bpermute(((lane ^ 16) << 2),
                                                       __float_as_int(v)));
}

// ---- 6 WMMAs of one hidden layer: Z^T = W^T * H^T (+bias via C) for 3 JVP streams
#define GEMM6(A0, A1, BC0, BC1)                                                                               \
    {                                                                                                         \
        v8f zz = {0.f, 0.f, 0.f, 0.f, 0.f, 0.f, 0.f, 0.f};                                                    \
        D[0][0] = __builtin_amdgcn_wmma_f32_16x16x32_f16(false, A0.h, false, Bh.h,  (short)0, BC0, false, false); \
        D[0][1] = __builtin_amdgcn_wmma_f32_16x16x32_f16(false, A1.h, false, Bh.h,  (short)0, BC1, false, false); \
        D[1][0] = __builtin_amdgcn_wmma_f32_16x16x32_f16(false, A0.h, false, Bd1.h, (short)0, zz,  false, false); \
        D[1][1] = __builtin_amdgcn_wmma_f32_16x16x32_f16(false, A1.h, false, Bd1.h, (short)0, zz,  false, false); \
        D[2][0] = __builtin_amdgcn_wmma_f32_16x16x32_f16(false, A0.h, false, Bd2.h, (short)0, zz,  false, false); \
        D[2][1] = __builtin_amdgcn_wmma_f32_16x16x32_f16(false, A1.h, false, Bd2.h, (short)0, zz,  false, false); \
    }

// ---- tanh + JVP chain rule, elementwise in D layout
#define ACTDERIV()                                                                  \
    _Pragma("unroll") for (int t = 0; t < 2; ++t) {                                 \
        _Pragma("unroll") for (int g = 0; g < 8; ++g) {                             \
            float z = D[0][t][g], dz = D[1][t][g], d2z = D[2][t][g];                \
            float h = tanh_fast(z); float tp = 1.0f - h * h;                        \
            D[0][t][g] = h;                                                         \
            D[1][t][g] = tp * dz;                                                   \
            D[2][t][g] = fmaf(tp, d2z, -2.0f * h * tp * dz * dz);                   \
        }                                                                           \
    }

// ---- D layout (f32, 2 row-tiles) -> f16 B operand of next layer (xor-16 lane swap)
#define REPACK(Bdst, s)                                                             \
    {                                                                               \
        uint32_t P0[4], P1[4];                                                      \
        _Pragma("unroll") for (int j = 0; j < 4; ++j) {                             \
            P0[j] = pack2(D[s][0][2 * j], D[s][0][2 * j + 1]);                      \
            P1[j] = pack2(D[s][1][2 * j], D[s][1][2 * j + 1]);                      \
        }                                                                           \
        _Pragma("unroll") for (int j = 0; j < 4; ++j) {                             \
            uint32_t s0 = permx16_u(P0[j], lane);                                   \
            uint32_t s1 = permx16_u(P1[j], lane);                                   \
            Bdst.u[j]     = lo ? P0[j] : s1;                                        \
            Bdst.u[4 + j] = lo ? s0 : P1[j];                                        \
        }                                                                           \
    }

__global__ __launch_bounds__(256) void pinn_mlp_jvp2_wmma(
    const float* __restrict__ X,
    const float* __restrict__ W1, const float* __restrict__ B1,
    const float* __restrict__ W2, const float* __restrict__ B2,
    const float* __restrict__ W3, const float* __restrict__ B3,
    const float* __restrict__ W4, const float* __restrict__ B4,
    float* __restrict__ out, int n)
{
    const int lane  = threadIdx.x & 31;
    const int wave  = blockIdx.x * (blockDim.x >> 5) + (threadIdx.x >> 5);
    const int nWav  = gridDim.x * (blockDim.x >> 5);
    const int tiles = n >> 4;

    const bool lo  = lane < 16;
    const int col  = lane & 15;
    const int klo  = lo ? 0 : 8;   // unused; A and B both take K base from kb1
    const int rlo  = lo ? 0 : 8;   // D-layout row base
    const int kb1  = lane & 16;    // B-layout K base (layer-1 direct build)

    // ---- one-time per-wave register preload of weights in WMMA operand layouts
    AV a2[2], a3[2];               // A = W^T row-blocks (f16)
#pragma unroll
    for (int e = 0; e < 16; ++e) {
        int k = kb1 + e;                               // A element e -> K index
        a2[0].h[e] = (_Float16)W2[k * 32 + col];
        a2[1].h[e] = (_Float16)W2[k * 32 + 16 + col];
        a3[0].h[e] = (_Float16)W3[k * 32 + col];
        a3[1].h[e] = (_Float16)W3[k * 32 + 16 + col];
    }
    v8f bc2[2], bc3[2];            // biases as C operand (D layout)
    float w4v[2][8];
#pragma unroll
    for (int g = 0; g < 8; ++g) {
        bc2[0][g] = B2[rlo + g];       bc2[1][g] = B2[16 + rlo + g];
        bc3[0][g] = B3[rlo + g];       bc3[1][g] = B3[16 + rlo + g];
        w4v[0][g] = W4[rlo + g];       w4v[1][g] = W4[16 + rlo + g];
    }
    float w1v[16], b1v[16];
#pragma unroll
    for (int e = 0; e < 16; ++e) { w1v[e] = W1[kb1 + e]; b1v[e] = B1[kb1 + e]; }
    const float b4 = B4[0];

    for (int tile = wave; tile < tiles; tile += nWav) {
        const int p0 = tile << 4;
        int pidx = p0 + col; if (pidx >= n) pidx = n - 1;
        const float xp = X[pidx];

        // ---- layer 1 (outer product; dz=W1, d2z=0) built directly in B layout
        union BV { v16h h; uint32_t u[8]; } Bh, Bd1, Bd2;
#pragma unroll
        for (int j = 0; j < 8; ++j) {
            float hv[2], d1v[2], d2v[2];
#pragma unroll
            for (int q = 0; q < 2; ++q) {
                int e = 2 * j + q;
                float w = w1v[e];
                float z = fmaf(xp, w, b1v[e]);
                float h = tanh_fast(z); float tp = 1.0f - h * h;
                hv[q] = h; d1v[q] = tp * w; d2v[q] = -2.0f * h * tp * w * w;
            }
            Bh.u[j]  = pack2(hv[0],  hv[1]);
            Bd1.u[j] = pack2(d1v[0], d1v[1]);
            Bd2.u[j] = pack2(d2v[0], d2v[1]);
        }

        v8f D[3][2];   // [stream: val,d1,d2][row-tile 0..1]

        // ---- layer 2 (WMMA) + activation + repack D->B
        GEMM6(a2[0], a2[1], bc2[0], bc2[1]);
        ACTDERIV();
        REPACK(Bh, 0); REPACK(Bd1, 1); REPACK(Bd2, 2);

        // ---- layer 3 (WMMA) + activation (stays in D layout)
        GEMM6(a3[0], a3[1], bc3[0], bc3[1]);
        ACTDERIV();

        // ---- layer 4: three dot products with W4, xor-16 partner reduction
        float accY = 0.f, acc1 = 0.f, acc2 = 0.f;
#pragma unroll
        for (int t = 0; t < 2; ++t)
#pragma unroll
            for (int g = 0; g < 8; ++g) {
                accY = fmaf(D[0][t][g], w4v[t][g], accY);
                acc1 = fmaf(D[1][t][g], w4v[t][g], acc1);
                acc2 = fmaf(D[2][t][g], w4v[t][g], acc2);
            }
        accY += permx16_f(accY, lane);
        acc1 += permx16_f(acc1, lane);
        acc2 += permx16_f(acc2, lane);

        const int p = p0 + col;
        if (lo && p < n) {
            out[2 + p] = acc2;                       // d2y/dx2, full [N]
            if (p == 0) { out[0] = accY + b4; out[1] = acc1; }  // y[0], dy[0]
        }
    }
}

extern "C" void kernel_launch(void* const* d_in, const int* in_sizes, int n_in,
                              void* d_out, int out_size, void* d_ws, size_t ws_size,
                              hipStream_t stream) {
    const float* X  = (const float*)d_in[0];
    const float* W1 = (const float*)d_in[1];
    const float* B1 = (const float*)d_in[2];
    const float* W2 = (const float*)d_in[3];
    const float* B2 = (const float*)d_in[4];
    const float* W3 = (const float*)d_in[5];
    const float* B3 = (const float*)d_in[6];
    const float* W4 = (const float*)d_in[7];
    const float* B4 = (const float*)d_in[8];
    float* out = (float*)d_out;
    const int n = in_sizes[0];

    const int tiles = n >> 4;
    const int wavesPerBlock = 256 / 32;
    int blocks = (tiles + wavesPerBlock * 8 - 1) / (wavesPerBlock * 8);  // ~8 tiles/wave
    if (blocks > 1024) blocks = 1024;
    if (blocks < 1)    blocks = 1;

    pinn_mlp_jvp2_wmma<<<blocks, 256, 0, stream>>>(X, W1, B1, W2, B2, W3, B3, W4, B4, out, n);
}
